// MPNNEdgePredictor_73108933312940
// MI455X (gfx1250) — matrix operand order repaired and checked by reading, not compiled, not run on hardware
//
#include <hip/hip_runtime.h>

// ---------------------------------------------------------------------------
// MPNN edge predictor for gfx1250 (MI455X), f16 WMMA path, 64-row M-tiles.
// G=2048 graphs, 57 nodes/graph, 160 edges/graph, HID=256, L=5.
// ---------------------------------------------------------------------------

#define HID 256
#define HID2 512
constexpr int G_      = 2048;
constexpr int NPG_    = 57;
constexpr int L_      = 5;
constexpr int NN_     = G_ * NPG_;    // 116736 nodes
constexpr int NE_     = G_ * 160;     // 327680 edges
constexpr int NBR_    = G_ * 80;      // 163840 branch pairs

typedef __attribute__((ext_vector_type(16))) _Float16 v16h;
typedef __attribute__((ext_vector_type(8)))  _Float16 v8h;
typedef __attribute__((ext_vector_type(8)))  float    v8f;

// ---- WMMA fragment helpers -------------------------------------------------
// A 16x32 f16 fragment from a row-major f16 tile in LDS (row stride `lda`):
//  lanes 0-15 : row = lane,     elems 0-7 -> K=k0+0..7,  elems 8-15 -> K=k0+16..23
//  lanes16-31 : row = lane-16,  elems 0-7 -> K=k0+8..15, elems 8-15 -> K=k0+24..31
__device__ __forceinline__ v16h a_frag(const _Float16* sA, int lda, int k0, int lane) {
    int r  = lane & 15;
    int kh = (lane & 16) ? 8 : 0;
    const _Float16* p = sA + r * lda + k0 + kh;
    v16h a;
    ((v8h*)&a)[0] = *(const v8h*)(p);        // K = k0+kh    .. +7
    ((v8h*)&a)[1] = *(const v8h*)(p + 16);   // K = k0+16+kh .. +7
    return a;
}

// B 32x16 f16 fragment from row-major [K x HID] f16 weights in global memory:
// lane L holds column n0+(L&15); elems 0-7 -> K=k0+kh+0..7, elems 8-15 -> K=k0+16+kh+0..7 (kh = 8*(L>>4)).
__device__ __forceinline__ v16h b_frag(const _Float16* W, int k0, int n0, int lane) {
    const _Float16* p = W + (size_t)(k0 + ((lane & 16) ? 8 : 0)) * HID + n0 + (lane & 15);
    v16h b;
    #pragma unroll
    for (int s = 0; s < 8; s++) {
        b[s]     = p[(size_t)s * HID];
        b[s + 8] = p[(size_t)(16 + s) * HID];
    }
    return b;
}

__device__ __forceinline__ v8f wmma(v16h a, v16h b, v8f c) {
    return __builtin_amdgcn_wmma_f32_16x16x32_f16(false, a, false, b, (short)0, c,
                                                  false, false);
}

// 4 row-tiles x 2 col-tiles of C (64 x 32 per wave) over K = KTOT.
// Each B fragment pair is reused by 4 WMMAs -> 4x L2 weight-traffic reduction.
template <int KTOT, int LDA>
__device__ __forceinline__ void gemm4x2(const _Float16* sA, const _Float16* W,
                                        int n0, int lane, v8f (&acc)[4][2]) {
    #pragma unroll
    for (int k0 = 0; k0 < KTOT; k0 += 32) {
        v16h b0 = b_frag(W, k0, n0, lane);
        v16h b1 = b_frag(W, k0, n0 + 16, lane);
        #pragma unroll
        for (int m = 0; m < 4; m++) {
            v16h a = a_frag(sA + m * 16 * LDA, LDA, k0, lane);
            acc[m][0] = wmma(a, b0, acc[m][0]);
            acc[m][1] = wmma(a, b1, acc[m][1]);
        }
    }
}

// ---- small utility kernels -------------------------------------------------
__global__ __launch_bounds__(256) void k_cvt_f16(const float* __restrict__ s,
                                                 _Float16* __restrict__ d, int n) {
    int i = blockIdx.x * 256 + threadIdx.x;
    if (i < n) d[i] = (_Float16)s[i];
}

// h = x @ in_w + in_b  ([NN,16] @ [16,256]); one thread per output element.
__global__ __launch_bounds__(256) void k_proj(const float* __restrict__ x,
                                              const float* __restrict__ w,
                                              const float* __restrict__ b,
                                              float* __restrict__ h32,
                                              _Float16* __restrict__ h16) {
    size_t gid = (size_t)blockIdx.x * 256 + threadIdx.x;
    int n = (int)(gid >> 8);
    int c = (int)(gid & 255);
    const float* xr = x + (size_t)n * 16;
    float s = b[c];
    #pragma unroll
    for (int k = 0; k < 16; k++) s += xr[k] * w[k * HID + c];
    h32[gid] = s;
    h16[gid] = (_Float16)s;
}

__global__ __launch_bounds__(256) void k_deg(const int* __restrict__ ei,
                                             float* __restrict__ deg, int E) {
    int e = blockIdx.x * 256 + threadIdx.x;
    if (e < E) atomicAdd(&deg[ei[E + e]], 1.0f);
}

// ---- per-layer edge message kernel ----------------------------------------
// block = 256 threads (8 waves), tile = 64 edges x 256 outputs.
// m = ReLU(cat[h[dst],h[src]] @ w1 + b1) @ w2 + b2 ; scatter-add into agg[dst].
__global__ __launch_bounds__(256) void k_edge(const _Float16* __restrict__ h16,
                                              const int* __restrict__ ei,
                                              const _Float16* __restrict__ w1,
                                              const float* __restrict__ b1,
                                              const _Float16* __restrict__ w2,
                                              const float* __restrict__ b2,
                                              float* __restrict__ agg) {
    __shared__ __align__(32) _Float16 sA1[64 * HID2];  // 64 KB: cat[h_dst,h_src]
    __shared__ __align__(32) _Float16 sA2[64 * HID];   // 32 KB: GEMM1 output
    __shared__ int sDst[64];

    const int tid = threadIdx.x;
    const int e0  = blockIdx.x * 64;
    const int row = tid >> 2, part = tid & 3;   // 4 threads stage one 512-wide row
    const int e   = e0 + row;
    const int src = ei[e];
    const int dst = ei[NE_ + e];
    if (part == 0) sDst[row] = dst;

    // stage cat[h[dst], h[src]] : 128 halves (256B) contiguous per thread
    const _Float16* sp = (part < 2)
        ? h16 + (size_t)dst * HID + part * 128
        : h16 + (size_t)src * HID + (part - 2) * 128;
    int4*       dp = (int4*)&sA1[row * HID2 + part * 128];
    const int4* gp = (const int4*)sp;
    #pragma unroll
    for (int j = 0; j < 16; j++) dp[j] = gp[j];
    __syncthreads();

    const int wave = tid >> 5, lane = tid & 31;
    const int n0  = wave * 32;
    const int col = lane & 15;
    const int rb  = (lane >> 4) * 8;

    // GEMM1: [64x512] x [512x256] -> ReLU -> sA2 (f16)
    v8f c1t[4][2] = {};
    gemm4x2<HID2, HID2>(sA1, w1, n0, lane, c1t);
    #pragma unroll
    for (int m = 0; m < 4; m++)
        #pragma unroll
        for (int j = 0; j < 8; j++) {
            int r = m * 16 + rb + j;
            float v0 = c1t[m][0][j] + b1[n0 + col];
            float v1 = c1t[m][1][j] + b1[n0 + 16 + col];
            sA2[r * HID + n0 + col]      = (_Float16)fmaxf(v0, 0.0f);
            sA2[r * HID + n0 + 16 + col] = (_Float16)fmaxf(v1, 0.0f);
        }
    __syncthreads();

    // GEMM2: [64x256] x [256x256] + b2 ; scatter-add to agg[dst]
    v8f c2t[4][2] = {};
    gemm4x2<HID, HID>(sA2, w2, n0, lane, c2t);
    #pragma unroll
    for (int m = 0; m < 4; m++)
        #pragma unroll
        for (int j = 0; j < 8; j++) {
            int r = m * 16 + rb + j;
            int dnode = sDst[r];
            float v0 = c2t[m][0][j] + b2[n0 + col];
            float v1 = c2t[m][1][j] + b2[n0 + 16 + col];
            atomicAdd(&agg[(size_t)dnode * HID + n0 + col], v0);
            atomicAdd(&agg[(size_t)dnode * HID + n0 + 16 + col], v1);
        }
}

// ---- per-layer node update kernel -----------------------------------------
// tile = 64 nodes; u = ReLU(BN(MLP2(cat[h, agg/deg]))) + h ; rewrites h32/h16.
__global__ __launch_bounds__(256) void k_update(float* __restrict__ h32,
                                                _Float16* __restrict__ h16,
                                                const float* __restrict__ agg,
                                                const float* __restrict__ deg,
                                                const _Float16* __restrict__ w1,
                                                const float* __restrict__ b1,
                                                const _Float16* __restrict__ w2,
                                                const float* __restrict__ b2,
                                                const float* __restrict__ bng,
                                                const float* __restrict__ bnb,
                                                const float* __restrict__ bnm,
                                                const float* __restrict__ bnv) {
    __shared__ __align__(32) _Float16 sA1[64 * HID2];
    __shared__ __align__(32) _Float16 sA2[64 * HID];

    const int tid  = threadIdx.x;
    const int base = blockIdx.x * 64;
    const int row  = tid >> 2, part = tid & 3;
    const int node = base + row;

    if (part < 2) {  // h half : direct f16 copy (256B contiguous)
        const int4* gp = (const int4*)(h16 + (size_t)node * HID + part * 128);
        int4* dp = (int4*)&sA1[row * HID2 + part * 128];
        #pragma unroll
        for (int j = 0; j < 16; j++) dp[j] = gp[j];
    } else {         // agg half : f32 -> (/deg) -> f16, packed 128-bit LDS stores
        float sc = 1.0f / fmaxf(deg[node], 1.0f);
        const float4* ap =
            (const float4*)(agg + (size_t)node * HID + (part - 2) * 128);
        int4* dp = (int4*)&sA1[row * HID2 + HID + (part - 2) * 128];
        #pragma unroll
        for (int j = 0; j < 16; j++) {
            float4 f0 = ap[2 * j];
            float4 f1 = ap[2 * j + 1];
            v8h hv;
            hv[0] = (_Float16)(f0.x * sc); hv[1] = (_Float16)(f0.y * sc);
            hv[2] = (_Float16)(f0.z * sc); hv[3] = (_Float16)(f0.w * sc);
            hv[4] = (_Float16)(f1.x * sc); hv[5] = (_Float16)(f1.y * sc);
            hv[6] = (_Float16)(f1.z * sc); hv[7] = (_Float16)(f1.w * sc);
            dp[j] = *(const int4*)&hv;
        }
    }
    __syncthreads();

    const int wave = tid >> 5, lane = tid & 31;
    const int n0  = wave * 32;
    const int col = lane & 15;
    const int rb  = (lane >> 4) * 8;

    v8f c1t[4][2] = {};
    gemm4x2<HID2, HID2>(sA1, w1, n0, lane, c1t);
    #pragma unroll
    for (int m = 0; m < 4; m++)
        #pragma unroll
        for (int j = 0; j < 8; j++) {
            int r = m * 16 + rb + j;
            float v0 = c1t[m][0][j] + b1[n0 + col];
            float v1 = c1t[m][1][j] + b1[n0 + 16 + col];
            sA2[r * HID + n0 + col]      = (_Float16)fmaxf(v0, 0.0f);
            sA2[r * HID + n0 + 16 + col] = (_Float16)fmaxf(v1, 0.0f);
        }
    __syncthreads();

    v8f c2t[4][2] = {};
    gemm4x2<HID, HID>(sA2, w2, n0, lane, c2t);
    #pragma unroll
    for (int m = 0; m < 4; m++)
        #pragma unroll
        for (int j = 0; j < 8; j++) {
            int nd = base + m * 16 + rb + j;
            int cA = n0 + col, cB = n0 + 16 + col;
            float u0 = c2t[m][0][j] + b2[cA];
            float u1 = c2t[m][1][j] + b2[cB];
            u0 = (u0 - bnm[cA]) * rsqrtf(bnv[cA] + 1e-5f) * bng[cA] + bnb[cA];
            u1 = (u1 - bnm[cB]) * rsqrtf(bnv[cB] + 1e-5f) * bng[cB] + bnb[cB];
            size_t iA = (size_t)nd * HID + cA, iB = (size_t)nd * HID + cB;
            float o0 = fmaxf(u0, 0.0f) + h32[iA];
            float o1 = fmaxf(u1, 0.0f) + h32[iB];
            h32[iA] = o0; h16[iA] = (_Float16)o0;
            h32[iB] = o1; h16[iB] = (_Float16)o1;
        }
}

// ---- final branch-pair MLP -------------------------------------------------
// tile = 64 branch rows; ef = cat[h[u],h[v]]; out = ReLU(ef@w1+b1)@w2+b2.
__global__ __launch_bounds__(256) void k_final(const _Float16* __restrict__ h16,
                                               const _Float16* __restrict__ w1,
                                               const float* __restrict__ b1,
                                               const float* __restrict__ w2,
                                               const float* __restrict__ b2,
                                               float* __restrict__ out) {
    __shared__ __align__(32) _Float16 sA[64 * HID2];   // 64 KB
    __shared__ __align__(32) _Float16 sH[64 * HID];    // 32 KB hidden (f16)
    __shared__ float sW2[HID];                          // 1 KB w2 stage

    const int tid = threadIdx.x;
    const int rb0 = blockIdx.x * 64;
    const int row = tid >> 2, part = tid & 3;
    const int r   = rb0 + row;
    const int g   = r / 80, i = r % 80;
    const int u   = g * NPG_ + (i % 57);
    const int v   = g * NPG_ + ((i * 13 + 1) % 57);

    sW2[tid] = w2[tid];  // 256 threads stage the [256] projection vector once

    const _Float16* sp = (part < 2)
        ? h16 + (size_t)u * HID + part * 128
        : h16 + (size_t)v * HID + (part - 2) * 128;
    int4*       dp = (int4*)&sA[row * HID2 + part * 128];
    const int4* gp = (const int4*)sp;
    #pragma unroll
    for (int j = 0; j < 16; j++) dp[j] = gp[j];
    __syncthreads();

    const int wave = tid >> 5, lane = tid & 31;
    const int n0  = wave * 32;
    const int col = lane & 15;
    const int rb  = (lane >> 4) * 8;

    v8f c1t[4][2] = {};
    gemm4x2<HID2, HID2>(sA, w1, n0, lane, c1t);
    #pragma unroll
    for (int m = 0; m < 4; m++)
        #pragma unroll
        for (int j = 0; j < 8; j++) {
            int rr = m * 16 + rb + j;
            sH[rr * HID + n0 + col] =
                (_Float16)fmaxf(c1t[m][0][j] + b1[n0 + col], 0.0f);
            sH[rr * HID + n0 + 16 + col] =
                (_Float16)fmaxf(c1t[m][1][j] + b1[n0 + 16 + col], 0.0f);
        }
    __syncthreads();

    // [256 -> 1] projection: each wave reduces 8 rows; each lane reads one
    // contiguous 16-half (32B) chunk of the row, then 16-lane shuffle tree.
    #pragma unroll
    for (int jj = 0; jj < 4; jj++) {
        int r2 = wave * 8 + jj * 2 + (lane >> 4);
        int cb = (lane & 15) * 16;
        v16h hv = *(const v16h*)&sH[r2 * HID + cb];
        float s = 0.0f;
        #pragma unroll
        for (int t = 0; t < 16; t++) s += (float)hv[t] * sW2[cb + t];
        #pragma unroll
        for (int msk = 1; msk < 16; msk <<= 1) s += __shfl_xor(s, msk, 32);
        if ((lane & 15) == 0) out[rb0 + r2] = s + b2[0];
    }
}

// ---------------------------------------------------------------------------
extern "C" void kernel_launch(void* const* d_in, const int* in_sizes, int n_in,
                              void* d_out, int out_size, void* d_ws, size_t ws_size,
                              hipStream_t stream) {
    (void)in_sizes; (void)n_in; (void)out_size; (void)ws_size;

    const float* x      = (const float*)d_in[0];
    const int*   ei     = (const int*)d_in[1];
    const float* in_w   = (const float*)d_in[3];
    const float* in_b   = (const float*)d_in[4];
    const float* msg_w1 = (const float*)d_in[5];
    const float* msg_b1 = (const float*)d_in[6];
    const float* msg_w2 = (const float*)d_in[7];
    const float* msg_b2 = (const float*)d_in[8];
    const float* upd_w1 = (const float*)d_in[9];
    const float* upd_b1 = (const float*)d_in[10];
    const float* upd_w2 = (const float*)d_in[11];
    const float* upd_b2 = (const float*)d_in[12];
    const float* bn_g   = (const float*)d_in[13];
    const float* bn_b   = (const float*)d_in[14];
    const float* bn_m   = (const float*)d_in[15];
    const float* bn_v   = (const float*)d_in[16];
    const float* mlp_w1 = (const float*)d_in[17];
    const float* mlp_b1 = (const float*)d_in[18];
    const float* mlp_w2 = (const float*)d_in[19];
    const float* mlp_b2 = (const float*)d_in[20];
    float* out = (float*)d_out;

    // workspace layout (all offsets 256B-aligned)
    char* ws = (char*)d_ws;
    const size_t sz_h32 = (size_t)NN_ * HID * 4;
    const size_t sz_h16 = (size_t)NN_ * HID * 2;
    const size_t sz_agg = (size_t)NN_ * HID * 4;
    const size_t sz_deg = (size_t)NN_ * 4;
    float*    h32 = (float*)(ws);
    _Float16* h16 = (_Float16*)(ws + sz_h32);
    float*    agg = (float*)(ws + sz_h32 + sz_h16);
    float*    deg = (float*)(ws + sz_h32 + sz_h16 + sz_agg);
    _Float16* wf  = (_Float16*)(ws + sz_h32 + sz_h16 + sz_agg + sz_deg);

    const int nMW1 = L_ * HID2 * HID;   // 655360
    const int nMW2 = L_ * HID * HID;    // 327680
    _Float16* mw1 = wf;
    _Float16* mw2 = mw1 + nMW1;
    _Float16* uw1 = mw2 + nMW2;
    _Float16* uw2 = uw1 + nMW1;
    _Float16* fw1 = uw2 + nMW2;         // [512 x 256]

    // 1) weights -> f16 (deterministic each call)
    hipLaunchKernelGGL(k_cvt_f16, dim3((nMW1 + 255) / 256), dim3(256), 0, stream,
                       msg_w1, mw1, nMW1);
    hipLaunchKernelGGL(k_cvt_f16, dim3((nMW2 + 255) / 256), dim3(256), 0, stream,
                       msg_w2, mw2, nMW2);
    hipLaunchKernelGGL(k_cvt_f16, dim3((nMW1 + 255) / 256), dim3(256), 0, stream,
                       upd_w1, uw1, nMW1);
    hipLaunchKernelGGL(k_cvt_f16, dim3((nMW2 + 255) / 256), dim3(256), 0, stream,
                       upd_w2, uw2, nMW2);
    hipLaunchKernelGGL(k_cvt_f16, dim3((HID2 * HID + 255) / 256), dim3(256), 0, stream,
                       mlp_w1, fw1, HID2 * HID);

    // 2) input projection
    hipLaunchKernelGGL(k_proj, dim3(NN_), dim3(256), 0, stream, x, in_w, in_b, h32, h16);

    // 3) in-degree
    hipMemsetAsync(deg, 0, sz_deg, stream);
    hipLaunchKernelGGL(k_deg, dim3((NE_ + 255) / 256), dim3(256), 0, stream, ei, deg, NE_);

    // 4) message-passing layers
    for (int l = 0; l < L_; l++) {
        hipMemsetAsync(agg, 0, sz_agg, stream);
        hipLaunchKernelGGL(k_edge, dim3(NE_ / 64), dim3(256), 0, stream,
                           h16, ei,
                           mw1 + (size_t)l * HID2 * HID, msg_b1 + l * HID,
                           mw2 + (size_t)l * HID * HID,  msg_b2 + l * HID,
                           agg);
        hipLaunchKernelGGL(k_update, dim3(NN_ / 64), dim3(256), 0, stream,
                           h32, h16, agg, deg,
                           uw1 + (size_t)l * HID2 * HID, upd_b1 + l * HID,
                           uw2 + (size_t)l * HID * HID,  upd_b2 + l * HID,
                           bn_g + l * HID, bn_b + l * HID,
                           bn_m + l * HID, bn_v + l * HID);
    }

    // 5) branch-pair MLP head
    hipLaunchKernelGGL(k_final, dim3(NBR_ / 64), dim3(256), 0, stream,
                       h16, fw1, mlp_b1, mlp_w2, mlp_b2, out);
}
